// SentimentAnalysisModel_15590731285009
// MI455X (gfx1250) — matrix operand order repaired and checked by reading, not compiled, not run on hardware
//
#include <hip/hip_runtime.h>
#include <hip/hip_bf16.h>

typedef __attribute__((ext_vector_type(16))) __bf16 v16bf;
typedef __attribute__((ext_vector_type(8)))  float  v8f;

#define B_   128
#define T_   256
#define L_   16
#define DC_  100
#define HC_  100
#define DW_  300
#define H_   300

__device__ __forceinline__ unsigned short f2bf(float f) {
  unsigned int u = __float_as_uint(f);
  u += 0x7FFFu + ((u >> 16) & 1u);   // round-to-nearest-even
  return (unsigned short)(u >> 16);
}
__device__ __forceinline__ float sigm_(float x) { return 1.0f / (1.0f + __expf(-x)); }
__device__ __forceinline__ float tanh_(float x) {
  float xc = fminf(fmaxf(x, -15.0f), 15.0f);
  float e = __expf(2.0f * xc);
  return (e - 1.0f) / (e + 1.0f);
}

// A-fragment index helper (v_wmma_f32_16x16x32_bf16 A layout, wave32):
//   lane (0..31): M = lane&15, half = lane>>4
//   element e (0..15): k = 8*half + e        (e<8 : VGPR0-3)
//                      k = 8*half + e + 8    (e>=8: VGPR4-7, K block +16)
__device__ __forceinline__ int frag_k(int lane, int e) {
  int half = lane >> 4;
  return 8 * half + e + ((e < 8) ? 0 : 8);
}

union FragU { uint4 u[2]; v16bf v; };

// Branch-free dual-tile GEMM pass: two independent accumulator chains share one
// A fragment per kc so the scheduler can interleave v_wmma and hide hazards.
// NS = row stride of both Wb and Zs (in elements).
template <int KC, int NS>
__device__ __forceinline__ void gemm_dual(const unsigned short* __restrict__ AF,
                                          const unsigned short* __restrict__ Wb,
                                          int lane, int nt0, int nt1,
                                          float* __restrict__ Zs) {
  v8f acc0 = {0.f, 0.f, 0.f, 0.f, 0.f, 0.f, 0.f, 0.f};
  v8f acc1 = {0.f, 0.f, 0.f, 0.f, 0.f, 0.f, 0.f, 0.f};
  for (int kc = 0; kc < KC; ++kc) {
    FragU a, b0, b1;
    const uint4* ap = (const uint4*)&AF[(kc * 32 + lane) * 16];
    a.u[0] = ap[0]; a.u[1] = ap[1];
    for (int e = 0; e < 16; ++e) {
      const unsigned short* brow = &Wb[(kc * 32 + frag_k(lane, e)) * NS + (lane & 15)];
      b0.v[e] = __builtin_bit_cast(__bf16, brow[nt0 * 16]);
      b1.v[e] = __builtin_bit_cast(__bf16, brow[nt1 * 16]);
    }
    acc0 = __builtin_amdgcn_wmma_f32_16x16x32_bf16(false, a.v, false, b0.v,
                                                   (short)0, acc0, false, false);
    acc1 = __builtin_amdgcn_wmma_f32_16x16x32_bf16(false, a.v, false, b1.v,
                                                   (short)0, acc1, false, false);
  }
  int Mr = 8 * (lane >> 4), nl = lane & 15;
#pragma unroll
  for (int r = 0; r < 8; ++r) {
    Zs[(Mr + r) * NS + nt0 * 16 + nl] = acc0[r];
    Zs[(Mr + r) * NS + nt1 * 16 + nl] = acc1[r];
  }
}

template <int KC, int NS>
__device__ __forceinline__ void gemm_single(const unsigned short* __restrict__ AF,
                                            const unsigned short* __restrict__ Wb,
                                            int lane, int nt,
                                            float* __restrict__ Zs) {
  v8f acc = {0.f, 0.f, 0.f, 0.f, 0.f, 0.f, 0.f, 0.f};
  for (int kc = 0; kc < KC; ++kc) {
    FragU a, b;
    const uint4* ap = (const uint4*)&AF[(kc * 32 + lane) * 16];
    a.u[0] = ap[0]; a.u[1] = ap[1];
    for (int e = 0; e < 16; ++e)
      b.v[e] = __builtin_bit_cast(__bf16, Wb[(kc * 32 + frag_k(lane, e)) * NS + nt * 16 + (lane & 15)]);
    acc = __builtin_amdgcn_wmma_f32_16x16x32_bf16(false, a.v, false, b.v,
                                                  (short)0, acc, false, false);
  }
  int Mr = 8 * (lane >> 4), nl = lane & 15;
#pragma unroll
  for (int r = 0; r < 8; ++r) Zs[(Mr + r) * NS + nt * 16 + nl] = acc[r];
}

// ---------------- weight convert: f32 [K,N] -> bf16 [Kpad,N] (zero pad) -------------
__global__ void convert_pad_kernel(const float* __restrict__ src,
                                   unsigned short* __restrict__ dst,
                                   int K, int Kpad, int N) {
  int i = blockIdx.x * blockDim.x + threadIdx.x;
  int total = Kpad * N;
  if (i >= total) return;
  int k = i / N, n = i - k * N;
  dst[i] = (k < K) ? f2bf(src[k * N + n]) : (unsigned short)0;
}

// ---------------- persistent char BiLSTM ---------------------------------------------
// grid (2048, 2): x = 16-row block of B*T=32768 words, y = direction.
// Whole L=16 recurrence in-kernel; h/c live in LDS; final h -> char_rep.
__global__ __launch_bounds__(256)
void char_bilstm_kernel(const int* __restrict__ char_ids,      // [B*T, L]
                        const int* __restrict__ word_lengths,  // [B*T]
                        const float* __restrict__ char_emb,    // [NC+1, DC]
                        const unsigned short* __restrict__ Wb_fw, const float* __restrict__ b_fw,
                        const unsigned short* __restrict__ Wb_bw, const float* __restrict__ b_bw,
                        unsigned short* __restrict__ rep) {    // bf16 [B*T, 2*HC]
  __shared__ __align__(16) unsigned short AF[7 * 32 * 16];  // A tiles, fragment order
  __shared__ __align__(16) float Zs[16 * 400];              // full gate row-block
  __shared__ __align__(16) unsigned short hs[16 * HC_];     // recurrent h (bf16)
  __shared__ __align__(16) float cs[16 * HC_];              // recurrent c (f32)
  __shared__ int s_wl[16];                                  // per-row lengths
  __shared__ int s_cb[16];                                  // per-row char_emb base (t-dep)

  const int dir = blockIdx.y;
  const unsigned short* Wb = dir ? Wb_bw : Wb_fw;
  const float* bias        = dir ? b_bw  : b_fw;

  const int tid  = threadIdx.x;
  const int m0   = blockIdx.x * 16;
  const int wave = __builtin_amdgcn_readfirstlane(tid >> 5);  // scalar wave id
  const int lane = tid & 31;

  for (int i = tid; i < 16 * HC_; i += 256) { hs[i] = 0; cs[i] = 0.0f; }
  if (tid < 16) s_wl[tid] = word_lengths[m0 + tid];
  __syncthreads();

  for (int t = 0; t < L_; ++t) {
    // Per-row gather bases for this timestep.
    if (tid < 16) {
      int wl = s_wl[tid];
      int te = dir ? ((t < wl) ? (wl - 1 - t) : t) : t;
      s_cb[tid] = char_ids[(m0 + tid) * L_ + te] * DC_;
    }
    __syncthreads();

    // Stage A = [char_emb gather | h | pad] directly in WMMA fragment order.
    for (int i = tid; i < 7 * 32 * 16; i += 256) {
      int kc = i >> 9, rem = i & 511;
      int ln = rem >> 4, e = rem & 15;
      int M = ln & 15;
      int K = kc * 32 + frag_k(ln, e);
      unsigned short v = 0;
      if (K < DC_) {
        v = f2bf(char_emb[s_cb[M] + K]);
      } else if (K < DC_ + HC_) {
        v = hs[M * HC_ + (K - DC_)];
      }
      AF[i] = v;
    }
    __syncthreads();

    // 25 N-tiles over 8 waves: dual {w, w+8}, single w+16, single 24 (wave 0).
    gemm_dual<7, 400>(AF, Wb, lane, wave, wave + 8, Zs);
    gemm_single<7, 400>(AF, Wb, lane, wave + 16, Zs);
    if (wave == 0) gemm_single<7, 400>(AF, Wb, lane, 24, Zs);
    __syncthreads();

    // Fused LSTM cell (gate order i, j, f, o) with length mask; state in LDS.
    for (int i = tid; i < 16 * HC_; i += 256) {
      int r = i / HC_, j = i - r * HC_;
      if (t < s_wl[r]) {
        float zi = Zs[r * 400 + j]           + bias[j];
        float zj = Zs[r * 400 + j + HC_]     + bias[j + HC_];
        float zf = Zs[r * 400 + j + 2 * HC_] + bias[j + 2 * HC_];
        float zo = Zs[r * 400 + j + 3 * HC_] + bias[j + 3 * HC_];
        float cn = cs[i] * sigm_(zf + 1.0f) + sigm_(zi) * tanh_(zj);
        float hn = sigm_(zo) * tanh_(cn);
        cs[i] = cn;
        hs[i] = f2bf(hn);
      }
    }
    __syncthreads();
  }

  // Final h -> char_rep column block [dir*HC, (dir+1)*HC)
  for (int i = tid; i < 16 * HC_; i += 256) {
    int r = i / HC_, j = i - r * HC_;
    rep[(m0 + r) * (2 * HC_) + dir * HC_ + j] = hs[i];
  }
}

// ---------------- persistent word BiLSTM ---------------------------------------------
// grid (8, 2): x = 16-row block over B=128, y = direction. 512 threads = 16 waves.
// Whole T=256 recurrence in-kernel; h/c live in LDS; final c -> global.
__global__ __launch_bounds__(512)
void word_bilstm_kernel(const int* __restrict__ word_ids,      // [B, T]
                        const int* __restrict__ seq_len,       // [B]
                        const float* __restrict__ word_emb,    // [NW, DW]
                        const unsigned short* __restrict__ char_rep, // bf16 [B*T, 2HC]
                        const unsigned short* __restrict__ Wb_fw, const float* __restrict__ b_fw,
                        const unsigned short* __restrict__ Wb_bw, const float* __restrict__ b_bw,
                        float* __restrict__ c_out_fw,          // f32 [B, H]
                        float* __restrict__ c_out_bw) {        // f32 [B, H]
  extern __shared__ __align__(16) char smem[];
  unsigned short* AF = (unsigned short*)smem;                   // 25*32*16 bf16 = 25600 B
  float* Zs          = (float*)(smem + 25600);                  // 16*1200 f32  = 76800 B
  unsigned short* hs = (unsigned short*)(smem + 25600 + 76800); // 16*300 bf16  =  9600 B
  float* cs          = (float*)(smem + 25600 + 76800 + 9600);   // 16*300 f32   = 19200 B
  int* s_len         = (int*)(smem + 131200);                   // 16 ints
  int* s_wb          = (int*)(smem + 131200 + 64);              // word_emb base per row
  int* s_rb          = (int*)(smem + 131200 + 128);             // char_rep base per row

  const int dir = blockIdx.y;
  const unsigned short* Wb = dir ? Wb_bw : Wb_fw;
  const float* bias        = dir ? b_bw  : b_fw;
  float* c_out             = dir ? c_out_bw : c_out_fw;

  const int tid  = threadIdx.x;
  const int m0   = blockIdx.x * 16;
  const int wave = __builtin_amdgcn_readfirstlane(tid >> 5);
  const int lane = tid & 31;

  for (int i = tid; i < 16 * H_; i += 512) { hs[i] = 0; cs[i] = 0.0f; }
  if (tid < 16) s_len[tid] = seq_len[m0 + tid];
  __syncthreads();

  for (int t = 0; t < T_; ++t) {
    if (tid < 16) {
      int len = s_len[tid];
      int te = dir ? ((t < len) ? (len - 1 - t) : t) : t;
      int m = m0 + tid;
      s_wb[tid] = word_ids[m * T_ + te] * DW_;
      s_rb[tid] = (m * T_ + te) * (2 * HC_);
    }
    __syncthreads();

    // Stage A = [word_emb gather | char_rep | h] in fragment order (K = 800).
    for (int i = tid; i < 25 * 32 * 16; i += 512) {
      int kc = i >> 9, rem = i & 511;
      int ln = rem >> 4, e = rem & 15;
      int M = ln & 15;
      int K = kc * 32 + frag_k(ln, e);
      unsigned short v;
      if (K < DW_) {
        v = f2bf(word_emb[s_wb[M] + K]);
      } else if (K < DW_ + 2 * HC_) {
        v = char_rep[s_rb[M] + (K - DW_)];
      } else {
        v = hs[M * H_ + (K - DW_ - 2 * HC_)];
      }
      AF[i] = v;
    }
    __syncthreads();

    // 75 N-tiles over 16 waves: dual {w, w+16}, dual {w+32, w+48}, single w+64.
    gemm_dual<25, 1200>(AF, Wb, lane, wave, wave + 16, Zs);
    gemm_dual<25, 1200>(AF, Wb, lane, wave + 32, wave + 48, Zs);
    if (wave < 11) gemm_single<25, 1200>(AF, Wb, lane, wave + 64, Zs);
    __syncthreads();

    for (int i = tid; i < 16 * H_; i += 512) {
      int r = i / H_, j = i - r * H_;
      if (t < s_len[r]) {
        float zi = Zs[r * 1200 + j]          + bias[j];
        float zj = Zs[r * 1200 + j + H_]     + bias[j + H_];
        float zf = Zs[r * 1200 + j + 2 * H_] + bias[j + 2 * H_];
        float zo = Zs[r * 1200 + j + 3 * H_] + bias[j + 3 * H_];
        float cn = cs[i] * sigm_(zf + 1.0f) + sigm_(zi) * tanh_(zj);
        float hn = sigm_(zo) * tanh_(cn);
        cs[i] = cn;
        hs[i] = f2bf(hn);
      }
    }
    __syncthreads();
  }

  // Final cell state -> global (reference uses state.c of last valid step).
  for (int i = tid; i < 16 * H_; i += 512) {
    int r = i / H_, j = i - r * H_;
    c_out[(m0 + r) * H_ + j] = cs[i];
  }
}

// ---------------- projection + log-softmax + mean NLL -------------------------------
__global__ void loss_kernel(const float* __restrict__ c_fw, const float* __restrict__ c_bw,
                            const float* __restrict__ W_proj, const float* __restrict__ b_proj,
                            const int* __restrict__ labels, float* __restrict__ out) {
  __shared__ float red[B_];
  int b = threadIdx.x;
  float lg0 = b_proj[0], lg1 = b_proj[1], lg2 = b_proj[2];
  for (int j = 0; j < H_; ++j) {
    float v = c_fw[b * H_ + j];
    lg0 += v * W_proj[j * 3 + 0];
    lg1 += v * W_proj[j * 3 + 1];
    lg2 += v * W_proj[j * 3 + 2];
  }
  for (int j = 0; j < H_; ++j) {
    float v = c_bw[b * H_ + j];
    lg0 += v * W_proj[(H_ + j) * 3 + 0];
    lg1 += v * W_proj[(H_ + j) * 3 + 1];
    lg2 += v * W_proj[(H_ + j) * 3 + 2];
  }
  float mx = fmaxf(lg0, fmaxf(lg1, lg2));
  float s = __expf(lg0 - mx) + __expf(lg1 - mx) + __expf(lg2 - mx);
  float lse = mx + __logf(s);
  int lab = labels[b];
  float lg = (lab == 0) ? lg0 : ((lab == 1) ? lg1 : lg2);
  red[b] = lse - lg;            // -logp[label]
  __syncthreads();
  for (int off = 64; off > 0; off >>= 1) {
    if (b < off) red[b] += red[b + off];
    __syncthreads();
  }
  if (b == 0) out[0] = red[0] / (float)B_;
}

extern "C" void kernel_launch(void* const* d_in, const int* in_sizes, int n_in,
                              void* d_out, int out_size, void* d_ws, size_t ws_size,
                              hipStream_t stream) {
  (void)in_sizes; (void)n_in; (void)out_size; (void)ws_size;

  const int*   word_ids = (const int*)d_in[0];
  const int*   seq_len  = (const int*)d_in[1];
  const int*   char_ids = (const int*)d_in[2];
  const int*   word_len = (const int*)d_in[3];
  const int*   labels   = (const int*)d_in[4];
  const float* word_emb = (const float*)d_in[5];
  const float* char_emb = (const float*)d_in[6];
  const float* Wc_fw    = (const float*)d_in[7];
  const float* bc_fw    = (const float*)d_in[8];
  const float* Wc_bw    = (const float*)d_in[9];
  const float* bc_bw    = (const float*)d_in[10];
  const float* Ww_fw    = (const float*)d_in[11];
  const float* bw_fw    = (const float*)d_in[12];
  const float* Ww_bw    = (const float*)d_in[13];
  const float* bw_bw    = (const float*)d_in[14];
  const float* W_proj   = (const float*)d_in[15];
  const float* b_proj   = (const float*)d_in[16];

  char* ws = (char*)d_ws;
  size_t off = 0;
#define WS_ALLOC(ty, name, bytes) \
  ty* name = (ty*)(ws + off); off = (off + (size_t)(bytes) + 255) & ~(size_t)255;

  WS_ALLOC(unsigned short, Wcfwb, 224 * 400 * 2)                 // char fw W bf16 (K-padded)
  WS_ALLOC(unsigned short, Wcbwb, 224 * 400 * 2)
  WS_ALLOC(unsigned short, Wwfwb, 800 * 1200 * 2)                // word fw W bf16
  WS_ALLOC(unsigned short, Wwbwb, 800 * 1200 * 2)
  WS_ALLOC(unsigned short, rep,   (size_t)B_ * T_ * 2 * HC_ * 2) // char_rep bf16 [B*T, 2HC]
  WS_ALLOC(float,          wc_fw, B_ * H_ * 4)                   // final word cell states
  WS_ALLOC(float,          wc_bw, B_ * H_ * 4)
#undef WS_ALLOC

  dim3 blk(256);
  convert_pad_kernel<<<(224 * 400 + 255) / 256, blk, 0, stream>>>(Wc_fw, Wcfwb, 200, 224, 400);
  convert_pad_kernel<<<(224 * 400 + 255) / 256, blk, 0, stream>>>(Wc_bw, Wcbwb, 200, 224, 400);
  convert_pad_kernel<<<(800 * 1200 + 255) / 256, blk, 0, stream>>>(Ww_fw, Wwfwb, 800, 800, 1200);
  convert_pad_kernel<<<(800 * 1200 + 255) / 256, blk, 0, stream>>>(Ww_bw, Wwbwb, 800, 800, 1200);

  // Entire char BiLSTM (both directions, all 16 steps) in one launch.
  char_bilstm_kernel<<<dim3((B_ * T_) / 16, 2), dim3(256), 0, stream>>>(
      char_ids, word_len, char_emb, Wcfwb, bc_fw, Wcbwb, bc_bw, rep);

  // Entire word BiLSTM (both directions, all 256 steps) in one launch.
  const size_t word_lds = 25600 + 76800 + 9600 + 19200 + 192;  // 131392 B
  word_bilstm_kernel<<<dim3(B_ / 16, 2), dim3(512), word_lds, stream>>>(
      word_ids, seq_len, word_emb, rep,
      Wwfwb, bw_fw, Wwbwb, bw_bw, wc_fw, wc_bw);

  loss_kernel<<<1, B_, 0, stream>>>(wc_fw, wc_bw, W_proj, b_proj, labels, (float*)d_out);
}
